// UniversalCalculator_15307263443373
// MI455X (gfx1250) — matrix operand order repaired and checked by reading, not compiled, not run on hardware
//
#include <hip/hip_runtime.h>

// ---------------------------------------------------------------------------
// MoE GLU MLP for MI455X (gfx1250).
//   - weights pre-transposed + converted to bf16 (one LDS-tiled pass)
//   - ALL GEMM tile staging via global_load_async_to_lds_b128 (ASYNCcnt)
//   - hot loop: ds_load_b128 fragments + v_wmma_f32_16x16x32_bf16 only
//   - deterministic combine (no float atomics)
// ---------------------------------------------------------------------------

#define D_MODEL     2048
#define H_EXP       1408
#define NUM_EXPERTS 8
#define NUM_TOKENS  8192
#define TOP_K       2
#define SLOTS       (NUM_TOKENS * TOP_K)   // 16384
#define CAP         (SLOTS / NUM_EXPERTS)  // 2048
#define SCORE_SCALE 1.0f

#define BM 128
#define BN 128
#define BK 32
#define LDST 40   // bf16 elements per LDS row (32 + 8 pad) = 80B; conflict-free b128 reads

typedef __attribute__((ext_vector_type(16))) __bf16 v16bf;
typedef __attribute__((ext_vector_type(8)))  __bf16 v8bf;
typedef __attribute__((ext_vector_type(4)))  __bf16 v4bf;
typedef __attribute__((ext_vector_type(8)))  float  v8f;
typedef __attribute__((ext_vector_type(4)))  float  v4f;

__device__ __forceinline__ v8f vzero8() {
  v8f z;
#pragma unroll
  for (int i = 0; i < 8; ++i) z[i] = 0.0f;
  return z;
}

__device__ __forceinline__ v8f wmma_bf16(v16bf a, v16bf b, v8f c) {
  // D = A(16x32 bf16) * B(32x16 bf16) + C(16x16 f32)
  return __builtin_amdgcn_wmma_f32_16x16x32_bf16(
      /*neg_a=*/false, a, /*neg_b=*/false, b,
      /*c_mod=*/(short)0, c, /*reuse_a=*/false, /*reuse_b=*/false);
}

// CDNA5 async 16B copy: global memory -> LDS, tracked by ASYNCcnt.
__device__ __forceinline__ void async_ld_b128(unsigned lds_addr, const void* gptr) {
  asm volatile("global_load_async_to_lds_b128 %0, %1, off"
               :: "v"(lds_addr), "v"(gptr) : "memory");
}
__device__ __forceinline__ void wait_async0() {
  asm volatile("s_wait_asynccnt 0" ::: "memory");
}

// A fragment: 16x32 bf16; lane L<16 -> row L, K {0..7,16..23}; L>=16 -> row L-16, K {8..15,24..31}
__device__ __forceinline__ v16bf ld_afrag(const __bf16* base, int lane, int row0) {
  int row = row0 + (lane & 15);
  const __bf16* p = base + row * LDST + ((lane >> 4) << 3);
  union { v16bf v; v8bf h[2]; } u;
  u.h[0] = *(const v8bf*)(p);
  u.h[1] = *(const v8bf*)(p + 16);
  return u.v;
}

// B fragment: 32x16 bf16 stored column-major ([N][K]); lane n<16 -> col n K {0..7,16..23}; n>=16 -> col n-16 K {8..15,24..31}
__device__ __forceinline__ v16bf ld_bfrag(const __bf16* base, int lane, int col0) {
  int col = col0 + (lane & 15);
  const __bf16* p = base + col * LDST + ((lane >> 4) << 3);
  union { v16bf v; v8bf h[2]; } u;
  u.h[0] = *(const v8bf*)(p);
  u.h[1] = *(const v8bf*)(p + 16);
  return u.v;
}

// ---------------------------------------------------------------------------
// Dispatch: stable counting sort of slots by expert (balanced -> CAP each).
// ---------------------------------------------------------------------------
__global__ __launch_bounds__(256) void moe_dispatch(
    const int* __restrict__ topk_idx, const float* __restrict__ topk_scores,
    int* __restrict__ sorted_batch, float* __restrict__ sorted_scores,
    int* __restrict__ slot_pos) {
  __shared__ int cnt[256][NUM_EXPERTS + 1];
  const int t = threadIdx.x;
#pragma unroll
  for (int e = 0; e < NUM_EXPERTS; ++e) cnt[t][e] = 0;
  __syncthreads();
  const int s0 = t * (SLOTS / 256);
  for (int i = 0; i < SLOTS / 256; ++i) cnt[t][topk_idx[s0 + i]]++;
  __syncthreads();
  if (t < NUM_EXPERTS) {  // exclusive scan over chunks for expert t
    int run = 0;
    for (int c = 0; c < 256; ++c) { int v = cnt[c][t]; cnt[c][t] = run; run += v; }
  }
  __syncthreads();
  for (int i = 0; i < SLOTS / 256; ++i) {
    int s = s0 + i;
    int e = topk_idx[s];
    int p = e * CAP + cnt[t][e]++;
    sorted_batch[p]  = s / TOP_K;
    sorted_scores[p] = topk_scores[s] * SCORE_SCALE;
    slot_pos[s]      = p;
  }
}

// ---------------------------------------------------------------------------
// Gather + convert: Xg[p, :] = bf16(x[sorted_batch[p], :]).  One block per slot.
// ---------------------------------------------------------------------------
__global__ __launch_bounds__(256) void moe_gather_x(
    const float* __restrict__ x, const int* __restrict__ sorted_batch,
    __bf16* __restrict__ Xg) {
  const int p = blockIdx.x;
  const int tok = sorted_batch[p];
  const float* src = x + (size_t)tok * D_MODEL;
  __bf16* dst = Xg + (size_t)p * D_MODEL;
  const int t = threadIdx.x;           // 256 threads x 8 elems = 2048
  v4f a = *(const v4f*)(src + t * 8);
  v4f b = *(const v4f*)(src + t * 8 + 4);
  v8bf o;
#pragma unroll
  for (int j = 0; j < 4; ++j) { o[j] = (__bf16)a[j]; o[4 + j] = (__bf16)b[j]; }
  *(v8bf*)(dst + t * 8) = o;
}

// ---------------------------------------------------------------------------
// Weight transpose+convert: in fp32 [E][R][C] -> out bf16 [E][C][R].
// 64x64 tiles through LDS; coalesced in and out.
// ---------------------------------------------------------------------------
__global__ __launch_bounds__(256) void moe_wconv_t(
    const float* __restrict__ in, __bf16* __restrict__ out, int R, int C) {
  __shared__ __bf16 tile[64][72];  // +8 pad kills bank conflicts on transposed reads
  const int e  = blockIdx.z;
  const int c0 = blockIdx.x * 64;
  const int r0 = blockIdx.y * 64;
  const int t  = threadIdx.x;
  const float* src = in + ((size_t)e * R + r0) * C + c0;
#pragma unroll
  for (int i = 0; i < 4; ++i) {          // 64x64 fp32 in, 4 elems/thread/iter
    int lin = t + i * 256;
    int row = lin >> 4, c4 = (lin & 15) * 4;
    v4f v = *(const v4f*)(src + (size_t)row * C + c4);
#pragma unroll
    for (int j = 0; j < 4; ++j) tile[row][c4 + j] = (__bf16)v[j];
  }
  __syncthreads();
  __bf16* dst = out + ((size_t)e * C + c0) * R + r0;
#pragma unroll
  for (int i = 0; i < 4; ++i) {          // 64x64 bf16 out, transposed
    int lin = t + i * 256;
    int cc = lin >> 4, r4 = (lin & 15) * 4;
    v4bf o;
#pragma unroll
    for (int j = 0; j < 4; ++j) o[j] = tile[r4 + j][cc];
    *(v4bf*)(dst + (size_t)cc * R + r4) = o;
  }
}

// ---------------------------------------------------------------------------
// GEMM1 (fused gate+up): H = silu(Xg @ Wg) * (Xg @ Wu), bf16 out.
// All three tiles (A, Bg, Bu) staged via async-to-LDS; zero VALU staging.
// ---------------------------------------------------------------------------
__global__ __launch_bounds__(256) void moe_gemm1(
    const __bf16* __restrict__ Xg, const __bf16* __restrict__ WgT,
    const __bf16* __restrict__ WuT, __bf16* __restrict__ Hbuf) {
  __shared__ __align__(16) __bf16 As[2][BM * LDST];
  __shared__ __align__(16) __bf16 Bg[2][BN * LDST];
  __shared__ __align__(16) __bf16 Bu[2][BN * LDST];

  const int e = blockIdx.z;
  const int m0 = blockIdx.y * BM;
  const int n0 = blockIdx.x * BN;
  const int tid = threadIdx.x;
  const int lane = tid & 31;
  const int wv = tid >> 5;
  const int wm = wv & 3;   // 32-row slab
  const int wn = wv >> 2;  // 64-col slab

  // Unified staging map: every tile is 128 rows x 32 bf16, 2 x 16B chunks/thread
  const __bf16 *pa[2], *pg[2], *pu[2];
  unsigned la[2][2], lg[2][2], lu[2][2];
#pragma unroll
  for (int i = 0; i < 2; ++i) {
    int lin = tid + i * 256;
    int row = lin >> 2, c8 = lin & 3;
    pa[i] = Xg  + ((size_t)(e * CAP  + m0 + row)) * D_MODEL + c8 * 8;
    pg[i] = WgT + ((size_t)(e * H_EXP + n0 + row)) * D_MODEL + c8 * 8;
    pu[i] = WuT + ((size_t)(e * H_EXP + n0 + row)) * D_MODEL + c8 * 8;
#pragma unroll
    for (int s = 0; s < 2; ++s) {
      la[s][i] = (unsigned)(size_t)&As[s][row * LDST + c8 * 8];
      lg[s][i] = (unsigned)(size_t)&Bg[s][row * LDST + c8 * 8];
      lu[s][i] = (unsigned)(size_t)&Bu[s][row * LDST + c8 * 8];
    }
  }

  auto stage = [&](int kt, int s) {
    size_t ko = (size_t)kt * BK;
#pragma unroll
    for (int i = 0; i < 2; ++i) async_ld_b128(la[s][i], pa[i] + ko);
#pragma unroll
    for (int i = 0; i < 2; ++i) async_ld_b128(lg[s][i], pg[i] + ko);
#pragma unroll
    for (int i = 0; i < 2; ++i) async_ld_b128(lu[s][i], pu[i] + ko);
  };

  v8f ag[2][4], au[2][4];
#pragma unroll
  for (int mt = 0; mt < 2; ++mt)
#pragma unroll
    for (int nt = 0; nt < 4; ++nt) { ag[mt][nt] = vzero8(); au[mt][nt] = vzero8(); }

  stage(0, 0);
  const int KT = D_MODEL / BK;  // 64
  for (int kt = 0; kt < KT; ++kt) {
    wait_async0();
    __syncthreads();
    const int s = kt & 1;
    if (kt + 1 < KT) stage(kt + 1, (kt + 1) & 1);
    if (kt + 8 < KT) {
      size_t po = (size_t)(kt + 8) * BK;
#pragma unroll
      for (int i = 0; i < 2; ++i) {
        __builtin_prefetch(pg[i] + po, 0, 0);
        __builtin_prefetch(pu[i] + po, 0, 0);
      }
    }
    v16bf af[2], bgf[4], buf[4];
#pragma unroll
    for (int mt = 0; mt < 2; ++mt) af[mt] = ld_afrag(&As[s][0], lane, wm * 32 + mt * 16);
#pragma unroll
    for (int nt = 0; nt < 4; ++nt) {
      bgf[nt] = ld_bfrag(&Bg[s][0], lane, wn * 64 + nt * 16);
      buf[nt] = ld_bfrag(&Bu[s][0], lane, wn * 64 + nt * 16);
    }
#pragma unroll
    for (int nt = 0; nt < 4; ++nt)
#pragma unroll
      for (int mt = 0; mt < 2; ++mt) {
        ag[mt][nt] = wmma_bf16(af[mt], bgf[nt], ag[mt][nt]);
        au[mt][nt] = wmma_bf16(af[mt], buf[nt], au[mt][nt]);
      }
  }

  // Epilogue: silu(gate) * up -> bf16 Hbuf
#pragma unroll
  for (int mt = 0; mt < 2; ++mt) {
#pragma unroll
    for (int nt = 0; nt < 4; ++nt) {
      int n = n0 + wn * 64 + nt * 16 + (lane & 15);
#pragma unroll
      for (int v = 0; v < 8; ++v) {
        int m = m0 + wm * 32 + mt * 16 + ((lane >> 4) << 3) + v;
        float g = ag[mt][nt][v], u = au[mt][nt][v];
        float sg = g / (1.0f + __expf(-g));
        Hbuf[(size_t)(e * CAP + m) * H_EXP + n] = (__bf16)(sg * u);
      }
    }
  }
}

// ---------------------------------------------------------------------------
// GEMM2: down = (H @ Wd) * score, fp32 out to workspace.
// ---------------------------------------------------------------------------
__global__ __launch_bounds__(256) void moe_gemm2(
    const __bf16* __restrict__ Hbuf, const __bf16* __restrict__ WdT,
    const float* __restrict__ sorted_scores, float* __restrict__ down) {
  __shared__ __align__(16) __bf16 As[2][BM * LDST];
  __shared__ __align__(16) __bf16 Bs[2][BN * LDST];

  const int e = blockIdx.z;
  const int m0 = blockIdx.y * BM;
  const int n0 = blockIdx.x * BN;
  const int tid = threadIdx.x;
  const int lane = tid & 31;
  const int wv = tid >> 5;
  const int wm = wv & 3;
  const int wn = wv >> 2;

  const __bf16 *pa[2], *pb[2];
  unsigned la[2][2], lb[2][2];
#pragma unroll
  for (int i = 0; i < 2; ++i) {
    int lin = tid + i * 256;
    int row = lin >> 2, c8 = lin & 3;
    pa[i] = Hbuf + ((size_t)(e * CAP + m0 + row)) * H_EXP + c8 * 8;
    pb[i] = WdT + ((size_t)(e * D_MODEL + n0 + row)) * H_EXP + c8 * 8;
#pragma unroll
    for (int s = 0; s < 2; ++s) {
      la[s][i] = (unsigned)(size_t)&As[s][row * LDST + c8 * 8];
      lb[s][i] = (unsigned)(size_t)&Bs[s][row * LDST + c8 * 8];
    }
  }

  auto stage = [&](int kt, int s) {
    size_t ko = (size_t)kt * BK;
#pragma unroll
    for (int i = 0; i < 2; ++i) async_ld_b128(la[s][i], pa[i] + ko);
#pragma unroll
    for (int i = 0; i < 2; ++i) async_ld_b128(lb[s][i], pb[i] + ko);
  };

  v8f acc[2][4];
#pragma unroll
  for (int mt = 0; mt < 2; ++mt)
#pragma unroll
    for (int nt = 0; nt < 4; ++nt) acc[mt][nt] = vzero8();

  stage(0, 0);
  const int KT = H_EXP / BK;  // 44
  for (int kt = 0; kt < KT; ++kt) {
    wait_async0();
    __syncthreads();
    const int s = kt & 1;
    if (kt + 1 < KT) stage(kt + 1, (kt + 1) & 1);
    if (kt + 8 < KT) {
      size_t po = (size_t)(kt + 8) * BK;
#pragma unroll
      for (int i = 0; i < 2; ++i) __builtin_prefetch(pb[i] + po, 0, 0);
    }
    v16bf af[2], bf[4];
#pragma unroll
    for (int mt = 0; mt < 2; ++mt) af[mt] = ld_afrag(&As[s][0], lane, wm * 32 + mt * 16);
#pragma unroll
    for (int nt = 0; nt < 4; ++nt) bf[nt] = ld_bfrag(&Bs[s][0], lane, wn * 64 + nt * 16);
#pragma unroll
    for (int nt = 0; nt < 4; ++nt)
#pragma unroll
      for (int mt = 0; mt < 2; ++mt) acc[mt][nt] = wmma_bf16(af[mt], bf[nt], acc[mt][nt]);
  }

#pragma unroll
  for (int mt = 0; mt < 2; ++mt) {
#pragma unroll
    for (int nt = 0; nt < 4; ++nt) {
      int n = n0 + wn * 64 + nt * 16 + (lane & 15);
#pragma unroll
      for (int v = 0; v < 8; ++v) {
        int m = m0 + wm * 32 + mt * 16 + ((lane >> 4) << 3) + v;
        float sc = sorted_scores[e * CAP + m];
        down[(size_t)(e * CAP + m) * D_MODEL + n] = acc[mt][nt][v] * sc;
      }
    }
  }
}

// ---------------------------------------------------------------------------
// Combine: y[b] = down[pos(b,0)] + down[pos(b,1)]  (no atomics, deterministic)
// ---------------------------------------------------------------------------
__global__ __launch_bounds__(256) void moe_combine(
    const float* __restrict__ down, const int* __restrict__ slot_pos,
    float* __restrict__ y) {
  int idx = blockIdx.x * 256 + threadIdx.x;       // over NUM_TOKENS * D/4
  int b = idx >> 9;                               // D_MODEL/4 = 512
  int c = idx & 511;
  int p0 = slot_pos[2 * b], p1 = slot_pos[2 * b + 1];
  v4f a = *(const v4f*)(down + (size_t)p0 * D_MODEL + c * 4);
  v4f d = *(const v4f*)(down + (size_t)p1 * D_MODEL + c * 4);
  *(v4f*)(y + (size_t)b * D_MODEL + c * 4) = a + d;
}

// ---------------------------------------------------------------------------
extern "C" void kernel_launch(void* const* d_in, const int* in_sizes, int n_in,
                              void* d_out, int out_size, void* d_ws, size_t ws_size,
                              hipStream_t stream) {
  const float* x           = (const float*)d_in[0];
  const int*   topk_idx    = (const int*)d_in[1];
  const float* topk_scores = (const float*)d_in[2];
  const float* Wg          = (const float*)d_in[3];
  const float* Wu          = (const float*)d_in[4];
  const float* Wd          = (const float*)d_in[5];
  float* y = (float*)d_out;

  // Workspace layout (~304 MB), fully rewritten each call.
  // Xg (67 MB, dead after GEMM1) shares the region of down (134 MB, GEMM2+).
  const size_t WSZ = (size_t)NUM_EXPERTS * D_MODEL * H_EXP * 2;  // 46,137,344
  char* ws = (char*)d_ws;
  int*    sorted_batch  = (int*)(ws);
  float*  sorted_scores = (float*)(ws + (1 << 16));
  int*    slot_pos      = (int*)(ws + (2 << 16));
  __bf16* WgT           = (__bf16*)(ws + (3 << 16));
  __bf16* WuT           = (__bf16*)(ws + (3 << 16) + WSZ);
  __bf16* WdT           = (__bf16*)(ws + (3 << 16) + 2 * WSZ);
  __bf16* Hbuf          = (__bf16*)(ws + (3 << 16) + 3 * WSZ);          // 46 MB
  char*   xd            = ws + (3 << 16) + 3 * WSZ + (size_t)SLOTS * H_EXP * 2;
  __bf16* Xg            = (__bf16*)xd;                                  // 67 MB (phase 1)
  float*  down          = (float*)xd;                                   // 134 MB (phase 2)

  moe_dispatch<<<1, 256, 0, stream>>>(topk_idx, topk_scores,
                                      sorted_batch, sorted_scores, slot_pos);

  moe_gather_x<<<SLOTS, 256, 0, stream>>>(x, sorted_batch, Xg);

  // Wg/Wu: [E][D][H] -> [E][H][D]; Wd: [E][H][D] -> [E][D][H]
  dim3 gw1(H_EXP / 64, D_MODEL / 64, NUM_EXPERTS);   // (22,32,8)
  moe_wconv_t<<<gw1, 256, 0, stream>>>(Wg, WgT, D_MODEL, H_EXP);
  moe_wconv_t<<<gw1, 256, 0, stream>>>(Wu, WuT, D_MODEL, H_EXP);
  dim3 gw2(D_MODEL / 64, H_EXP / 64, NUM_EXPERTS);   // (32,22,8)
  moe_wconv_t<<<gw2, 256, 0, stream>>>(Wd, WdT, H_EXP, D_MODEL);

  dim3 g1(H_EXP / BN, CAP / BM, NUM_EXPERTS);   // (11,16,8)
  moe_gemm1<<<g1, 256, 0, stream>>>(Xg, WgT, WuT, Hbuf);

  dim3 g2(D_MODEL / BN, CAP / BM, NUM_EXPERTS); // (16,16,8)
  moe_gemm2<<<g2, 256, 0, stream>>>(Hbuf, WdT, sorted_scores, down);

  moe_combine<<<(NUM_TOKENS * (D_MODEL / 4)) / 256, 256, 0, stream>>>(down, slot_pos, y);
}
